// SRCMLayer_8211977470535
// MI455X (gfx1250) — compile-verified, matrix operand order checked by reading, not run on hardware
//
#include <hip/hip_runtime.h>
#include <hip/hip_bf16.h>
#include <math.h>

// ---------------------------------------------------------------------------
// Problem constants (from reference)
// ---------------------------------------------------------------------------
#define Bx    8
#define Hh    64
#define Ww    64
#define Lx    (Hh * Ww)        // 4096
#define Cx    256              // D_MODEL / OUT_DIM
#define DIx   512              // D_INNER
#define Ex    (2 * DIx)        // 1024
#define DSx   16               // D_STATE
#define DTRx  16               // DT_RANK
#define KXP   (DTRx + 2 * DSx) // 48
#define DCONV 4

typedef __attribute__((ext_vector_type(16))) _Float16 v16h;
typedef __attribute__((ext_vector_type(8)))  float    v8f;

// ---------------------------------------------------------------------------
// Helpers
// ---------------------------------------------------------------------------
__device__ __forceinline__ float silu_f(float x) {
    return x / (1.0f + __expf(-x));
}
__device__ __forceinline__ float softplus_f(float x) {
    return (x > 20.0f) ? x : log1pf(__expf(x));
}

// Async global->LDS copy of one 16B chunk (CDNA5 GLOBAL_LOAD_ASYNC_TO_LDS_B128,
// tracked by ASYNCcnt). lds_addr is a 32-bit LDS byte address in a VGPR.
__device__ __forceinline__ void async_ld_b128(unsigned lds_addr, const void* gaddr) {
    asm volatile("global_load_async_to_lds_b128 %0, %1, off"
                 :: "v"(lds_addr), "v"((unsigned long long)(uintptr_t)gaddr)
                 : "memory");
}

__device__ __forceinline__ void wait_async0() {
#if __has_builtin(__builtin_amdgcn_s_wait_asynccnt)
    __builtin_amdgcn_s_wait_asynccnt(0);
#else
    asm volatile("s_wait_asynccnt 0x0" ::: "memory");
#endif
}

// Stage W[n0 : n0+ROWS, 0:KK] (f16, row-major, ld = KK) into LDS with row
// pitch KK+16 halves (32B-aligned rows). All 256 threads cooperate; each
// issues its share of async b128 transfers, waits its own ASYNCcnt, then the
// workgroup barrier makes the tile visible to every wave.
template <int ROWS, int KK>
__device__ __forceinline__ void stage_w(const _Float16* __restrict__ W, int n0,
                                        _Float16* tile) {
    const int PITCH = KK + 16;
    const int CPR   = KK / 8;          // 16B chunks per row
    const int CH    = ROWS * CPR;
    unsigned base = (unsigned)(uintptr_t)tile;
    const char* wb = (const char*)(W + (size_t)n0 * KK);
    for (int c = threadIdx.x; c < CH; c += 256) {
        int r = c / CPR, j = c - r * CPR;
        unsigned la = base + (unsigned)(r * PITCH + j * 8) * 2u;
        async_ld_b128(la, wb + (size_t)r * KK * 2 + (size_t)j * 16);
    }
    wait_async0();
    __syncthreads();
}

// Wave-level WMMA K-loop: A (M x K) f16 row-major from global; W fragments
// from the LDS tile staged by stage_w (pitch = K + 16 halves).
// Computes NT tiles D[m0:m0+16, 16*t : 16*t+16] of the block's N-group.
template <int NT>
__device__ __forceinline__ void wmma_tile_lds(const _Float16* __restrict__ A, int lda,
                                              const _Float16* Wl, int K,
                                              int m0, v8f* acc) {
    const int pitch = K + 16;
    const int lane  = threadIdx.x & 31;
    const int row   = lane & 15;
    const int hi    = lane >> 4;
    const _Float16* ap = A + (size_t)(m0 + row) * lda + hi * 16;
    const _Float16* wp = Wl + row * pitch + hi * 16;
    for (int k0 = 0; k0 < K; k0 += 32) {
        v16h a = *(const v16h*)(ap + k0);
#pragma unroll
        for (int t = 0; t < NT; ++t) {
            v16h wv = *(const v16h*)(wp + (t * 16) * pitch + k0);
            acc[t] = __builtin_amdgcn_wmma_f32_16x16x32_f16(
                false, a, false, wv, (short)0, acc[t], false, false);
        }
    }
}

// Store one 16x16 f32 tile "transposed": out[(n0+n)*ldm + m0+m] = D[m][n].
// Per ISA C/D layout each lane holds 8 consecutive M of one N column ->
// contiguous 8-float store per lane.
__device__ __forceinline__ void store_tile_tr(float* __restrict__ out, int ldm,
                                              int m0, int n0, v8f a, float bias) {
    const int lane = threadIdx.x & 31;
    const int col  = lane & 15;
    const int rb   = (lane >> 4) * 8;
    float* p = out + (size_t)(n0 + col) * ldm + m0 + rb;
    *(float4*)(p + 0) = make_float4(a[0] + bias, a[1] + bias, a[2] + bias, a[3] + bias);
    *(float4*)(p + 4) = make_float4(a[4] + bias, a[5] + bias, a[6] + bias, a[7] + bias);
}

// Store one 16x16 f32 tile row-major: out[(m0+m)*ldn + n0+n] = D[m][n].
__device__ __forceinline__ void store_tile_rm(float* __restrict__ out, int ldn,
                                              int m0, int n0, v8f a) {
    const int lane = threadIdx.x & 31;
    const int col  = lane & 15;
    const int rb   = (lane >> 4) * 8;
    float* p = out + (size_t)(m0 + rb) * ldn + n0 + col;
#pragma unroll
    for (int r = 0; r < 8; ++r) p[(size_t)r * ldn] = a[r];
}

// ---------------------------------------------------------------------------
// Kernels
// ---------------------------------------------------------------------------

__global__ void cvt_f16_kernel(const float* __restrict__ src, _Float16* __restrict__ dst, int n) {
    int i = blockIdx.x * 256 + threadIdx.x;
    if (i < n) dst[i] = (_Float16)src[i];
}

// LayerNorm over C for x stored (B, C, L); outputs f16 (B, L, C).
__global__ void ln1_kernel(const float* __restrict__ x, const float* __restrict__ g,
                           const float* __restrict__ be, _Float16* __restrict__ xn) {
    const int bl = blockIdx.x;
    const int b = bl / Lx, l = bl % Lx;
    const int c = threadIdx.x;
    float v = x[((size_t)b * Cx + c) * Lx + l];
    __shared__ float s1[256], s2[256];
    s1[c] = v; s2[c] = v * v;
    __syncthreads();
    for (int s = 128; s > 0; s >>= 1) {
        if (c < s) { s1[c] += s1[c + s]; s2[c] += s2[c + s]; }
        __syncthreads();
    }
    float m   = s1[0] * (1.0f / Cx);
    float var = s2[0] * (1.0f / Cx) - m * m;
    float r   = rsqrtf(var + 1e-5f);
    xn[((size_t)b * Lx + l) * Cx + c] = (_Float16)((v - m) * r * g[c] + be[c]);
}

// xm = mamba_out(B,L,C) + skip * x(B,C,L); LayerNorm over C; out f16 (B,L,C).
__global__ void ln2_kernel(const float* __restrict__ mo, const float* __restrict__ x,
                           const float* __restrict__ ss, const float* __restrict__ g,
                           const float* __restrict__ be, _Float16* __restrict__ xm) {
    const int bl = blockIdx.x;
    const int b = bl / Lx, l = bl % Lx;
    const int c = threadIdx.x;
    float v = mo[((size_t)b * Lx + l) * Cx + c] + ss[0] * x[((size_t)b * Cx + c) * Lx + l];
    __shared__ float s1[256], s2[256];
    s1[c] = v; s2[c] = v * v;
    __syncthreads();
    for (int s = 128; s > 0; s >>= 1) {
        if (c < s) { s1[c] += s1[c + s]; s2[c] += s2[c + s]; }
        __syncthreads();
    }
    float m   = s1[0] * (1.0f / Cx);
    float var = s2[0] * (1.0f / Cx) - m * m;
    float r   = rsqrtf(var + 1e-5f);
    xm[((size_t)b * Lx + l) * Cx + c] = (_Float16)((v - m) * r * g[c] + be[c]);
}

// xz(B, 1024, L) = xn(B, L, 256) x in_proj_w(1024, 256)^T   [stored (e, l)]
__global__ void gemm_inproj_kernel(const _Float16* __restrict__ xn,
                                   const _Float16* __restrict__ w,
                                   float* __restrict__ xz) {
    __shared__ __align__(32) _Float16 wt[64 * (Cx + 16)];
    const int b  = blockIdx.z;
    const int m0 = (blockIdx.x * 8 + (threadIdx.x >> 5)) * 16;
    const int n0 = blockIdx.y * 64;
    stage_w<64, Cx>(w, n0, wt);
    v8f acc[4] = {};
    wmma_tile_lds<4>(xn + (size_t)b * Lx * Cx, Cx, wt, Cx, m0, acc);
    float* outb = xz + (size_t)b * Ex * Lx;
#pragma unroll
    for (int t = 0; t < 4; ++t) store_tile_tr(outb, Lx, m0, n0 + 16 * t, acc[t], 0.0f);
}

// x_dbl(B, 48, L) = xcT(B, L, 512) x x_proj_w(48, 512)^T   [stored (k, l)]
__global__ void gemm_xproj_kernel(const _Float16* __restrict__ xcT,
                                  const _Float16* __restrict__ w,
                                  float* __restrict__ xdbl) {
    __shared__ __align__(32) _Float16 wt[KXP * (DIx + 16)];
    const int b  = blockIdx.z;
    const int m0 = (blockIdx.x * 8 + (threadIdx.x >> 5)) * 16;
    stage_w<KXP, DIx>(w, 0, wt);
    v8f acc[3] = {};
    wmma_tile_lds<3>(xcT + (size_t)b * Lx * DIx, DIx, wt, DIx, m0, acc);
    float* outb = xdbl + (size_t)b * KXP * Lx;
#pragma unroll
    for (int t = 0; t < 3; ++t) store_tile_tr(outb, Lx, m0, 16 * t, acc[t], 0.0f);
}

// mamba_out(B, L, 256) = ycT(B, L, 512) x out_proj_w(256, 512)^T  [row-major]
__global__ void gemm_outproj_kernel(const _Float16* __restrict__ ycT,
                                    const _Float16* __restrict__ w,
                                    float* __restrict__ mo) {
    __shared__ __align__(32) _Float16 wt[64 * (DIx + 16)];
    const int b  = blockIdx.z;
    const int m0 = (blockIdx.x * 8 + (threadIdx.x >> 5)) * 16;
    const int n0 = blockIdx.y * 64;
    stage_w<64, DIx>(w, n0, wt);
    v8f acc[4] = {};
    wmma_tile_lds<4>(ycT + (size_t)b * Lx * DIx, DIx, wt, DIx, m0, acc);
    float* outb = mo + (size_t)b * Lx * Cx;
#pragma unroll
    for (int t = 0; t < 4; ++t) store_tile_rm(outb, Cx, m0, n0 + 16 * t, acc[t]);
}

// out(B, 256, L) = xm(B, L, 256) x proj_w(256, 256)^T + proj_b   [stored (o, l)]
__global__ void gemm_proj_kernel(const _Float16* __restrict__ xm,
                                 const _Float16* __restrict__ w,
                                 const float* __restrict__ bias,
                                 float* __restrict__ out) {
    __shared__ __align__(32) _Float16 wt[64 * (Cx + 16)];
    const int b  = blockIdx.z;
    const int m0 = (blockIdx.x * 8 + (threadIdx.x >> 5)) * 16;
    const int n0 = blockIdx.y * 64;
    stage_w<64, Cx>(w, n0, wt);
    v8f acc[4] = {};
    wmma_tile_lds<4>(xm + (size_t)b * Lx * Cx, Cx, wt, Cx, m0, acc);
    float* outb = out + (size_t)b * Cx * Lx;
    const int col = threadIdx.x & 15;
#pragma unroll
    for (int t = 0; t < 4; ++t)
        store_tile_tr(outb, Lx, m0, n0 + 16 * t, acc[t], bias[n0 + 16 * t + col]);
}

// Causal depthwise conv (width 4) + bias + SiLU on xi = xz[:, :512, :].
// rev=1 computes on the flipped sequence (outputs stay in flipped "scan" coords).
// Writes f32 (B, D, L) for the scan and f16 (B, L, D) for the x_proj GEMM.
__global__ void conv_silu_kernel(const float* __restrict__ xz,
                                 const float* __restrict__ cw,
                                 const float* __restrict__ cb,
                                 float* __restrict__ xc,
                                 _Float16* __restrict__ xcT, int rev) {
    const size_t i = (size_t)blockIdx.x * 256 + threadIdx.x;  // < B*DI*L
    const int l = (int)(i % Lx);
    const size_t t = i / Lx;
    const int d = (int)(t % DIx);
    const int b = (int)(t / DIx);
    const float* xi = xz + ((size_t)b * Ex + d) * Lx;
    float acc = cb[d];
#pragma unroll
    for (int j = 0; j < DCONV; ++j) {
        int src = rev ? (Lx - 1 - l + (DCONV - 1) - j) : (l - (DCONV - 1) + j);
        if (src >= 0 && src < Lx) acc += cw[d * DCONV + j] * xi[src];
    }
    float v = silu_f(acc);
    xc[((size_t)b * DIx + d) * Lx + l]  = v;
    xcT[((size_t)b * Lx + l) * DIx + d] = (_Float16)v;
}

// dt(B, D, L) = softplus(x_dbl[:, :16, :]^T dt_w + dt_b)
__global__ void dt_kernel(const float* __restrict__ xdbl,
                          const float* __restrict__ dtw,
                          const float* __restrict__ dtb,
                          float* __restrict__ dt) {
    const size_t i = (size_t)blockIdx.x * 256 + threadIdx.x;  // < B*DI*L
    const int l = (int)(i % Lx);
    const size_t t = i / Lx;
    const int d = (int)(t % DIx);
    const int b = (int)(t / DIx);
    float acc = dtb[d];
    const float* xd = xdbl + (size_t)b * KXP * Lx + l;
#pragma unroll
    for (int k = 0; k < DTRx; ++k) acc += xd[(size_t)k * Lx] * dtw[d * DTRx + k];
    dt[((size_t)b * DIx + d) * Lx + l] = softplus_f(acc);
}

// Selective scan. One wave = 2 channels x 16 states (lane = dloc*16 + n).
// Sequential over L; y reduced over the 16 state lanes via shuffles.
// Streams are prefetched ahead (global_prefetch_b8) to hide latency behind
// the serial v_exp_f32 chain.
__global__ void scan_kernel(const float* __restrict__ dt,
                            const float* __restrict__ xc,
                            const float* __restrict__ xdbl,
                            const float* __restrict__ A_log,
                            float* __restrict__ ys) {
    const int b    = blockIdx.y;
    const int lane = threadIdx.x & 31;
    const int wv   = threadIdx.x >> 5;
    const int n    = lane & 15;
    const int d    = blockIdx.x * 16 + wv * 2 + (lane >> 4);

    const float Aval = -__expf(A_log[d * DSx + n]);
    const float* dtp = dt   + ((size_t)b * DIx + d) * Lx;
    const float* xp  = xc   + ((size_t)b * DIx + d) * Lx;
    const float* bp  = xdbl + ((size_t)b * KXP + DTRx + n) * Lx;
    const float* cp  = xdbl + ((size_t)b * KXP + DTRx + DSx + n) * Lx;
    float* yp        = ys   + ((size_t)b * DIx + d) * Lx;

    float h = 0.0f;
    for (int l = 0; l < Lx; ++l) {
        if ((l & 63) == 0 && l + 256 < Lx) {
            __builtin_prefetch(dtp + l + 256, 0, 0);
            __builtin_prefetch(xp  + l + 256, 0, 0);
            __builtin_prefetch(bp  + l + 256, 0, 0);
            __builtin_prefetch(cp  + l + 256, 0, 0);
        }
        float dtv = dtp[l];
        float xv  = xp[l];
        float bv  = bp[l];
        float cv  = cp[l];
        h = __expf(dtv * Aval) * h + (dtv * xv) * bv;
        float p = h * cv;
        p += __shfl_xor(p, 1);
        p += __shfl_xor(p, 2);
        p += __shfl_xor(p, 4);
        p += __shfl_xor(p, 8);
        if (n == 0) yp[l] = p;
    }
}

// y = (ys_f + Dp*xc_f)*silu(z) + (flip(ys_r) + Dp_b*flip(xc_r))*silu(z);
// written as f16 (B, L, D) for the out_proj GEMM.
__global__ void combine_kernel(const float* __restrict__ ysf, const float* __restrict__ ysr,
                               const float* __restrict__ xcf, const float* __restrict__ xcr,
                               const float* __restrict__ xz,
                               const float* __restrict__ Dp,  const float* __restrict__ Dpb,
                               _Float16* __restrict__ ycT) {
    const size_t i = (size_t)blockIdx.x * 256 + threadIdx.x;  // < B*DI*L
    const int l = (int)(i % Lx);
    const size_t t = i / Lx;
    const int d = (int)(t % DIx);
    const int b = (int)(t / DIx);
    const size_t fi = ((size_t)b * DIx + d) * Lx + l;
    const size_t ri = ((size_t)b * DIx + d) * Lx + (Lx - 1 - l);
    float z  = xz[((size_t)b * Ex + DIx + d) * Lx + l];
    float sz = silu_f(z);
    float y  = (ysf[fi] + Dp[d] * xcf[fi]) * sz + (ysr[ri] + Dpb[d] * xcr[ri]) * sz;
    ycT[((size_t)b * Lx + l) * DIx + d] = (_Float16)y;
}

// ---------------------------------------------------------------------------
// Host launcher
// ---------------------------------------------------------------------------
extern "C" void kernel_launch(void* const* d_in, const int* in_sizes, int n_in,
                              void* d_out, int out_size, void* d_ws, size_t ws_size,
                              hipStream_t stream) {
    const float* x          = (const float*)d_in[0];
    const float* norm_g     = (const float*)d_in[1];
    const float* norm_b     = (const float*)d_in[2];
    const float* skip_scale = (const float*)d_in[3];
    const float* proj_w     = (const float*)d_in[4];
    const float* proj_b     = (const float*)d_in[5];
    const float* in_proj_w  = (const float*)d_in[6];
    const float* out_proj_w = (const float*)d_in[7];
    const float* conv_w     = (const float*)d_in[8];
    const float* conv_b     = (const float*)d_in[9];
    const float* x_proj_w   = (const float*)d_in[10];
    const float* dt_proj_w  = (const float*)d_in[11];
    const float* dt_proj_b  = (const float*)d_in[12];
    const float* A_log      = (const float*)d_in[13];
    const float* Dp         = (const float*)d_in[14];
    const float* conv_w_b   = (const float*)d_in[15];
    const float* conv_b_b   = (const float*)d_in[16];
    const float* x_proj_w_b = (const float*)d_in[17];
    const float* dt_proj_w_b= (const float*)d_in[18];
    const float* dt_proj_b_b= (const float*)d_in[19];
    const float* A_log_b    = (const float*)d_in[20];
    const float* Dp_b       = (const float*)d_in[21];
    float* out = (float*)d_out;

    // Workspace carve-out (256B aligned)
    char* cur = (char*)d_ws;
    auto alloc = [&](size_t bytes) -> void* {
        void* p = (void*)cur;
        cur += (bytes + 255) & ~(size_t)255;
        return p;
    };
    _Float16* w_in   = (_Float16*)alloc((size_t)Ex * Cx * 2);
    _Float16* w_xp   = (_Float16*)alloc((size_t)KXP * DIx * 2);
    _Float16* w_xpb  = (_Float16*)alloc((size_t)KXP * DIx * 2);
    _Float16* w_out  = (_Float16*)alloc((size_t)Cx * DIx * 2);
    _Float16* w_proj = (_Float16*)alloc((size_t)Cx * Cx * 2);
    _Float16* xn     = (_Float16*)alloc((size_t)Bx * Lx * Cx * 2);
    float*    xz     = (float*)   alloc((size_t)Bx * Ex * Lx * 4);
    float*    xc_f   = (float*)   alloc((size_t)Bx * DIx * Lx * 4);
    float*    xc_r   = (float*)   alloc((size_t)Bx * DIx * Lx * 4);
    _Float16* xcT_f  = (_Float16*)alloc((size_t)Bx * Lx * DIx * 2);
    _Float16* xcT_r  = (_Float16*)alloc((size_t)Bx * Lx * DIx * 2);
    float*    xdbl_f = (float*)   alloc((size_t)Bx * KXP * Lx * 4);
    float*    xdbl_r = (float*)   alloc((size_t)Bx * KXP * Lx * 4);
    float*    dt_f   = (float*)   alloc((size_t)Bx * DIx * Lx * 4);
    float*    dt_r   = (float*)   alloc((size_t)Bx * DIx * Lx * 4);
    float*    ys_f   = (float*)   alloc((size_t)Bx * DIx * Lx * 4);
    float*    ys_r   = (float*)   alloc((size_t)Bx * DIx * Lx * 4);
    _Float16* ycT    = (_Float16*)alloc((size_t)Bx * Lx * DIx * 2);
    float*    mo     = (float*)   alloc((size_t)Bx * Lx * Cx * 4);
    _Float16* xm     = (_Float16*)alloc((size_t)Bx * Lx * Cx * 2);

    // 1) weight conversions to f16
    auto cvt = [&](const float* s, _Float16* d, int n) {
        cvt_f16_kernel<<<(n + 255) / 256, 256, 0, stream>>>(s, d, n);
    };
    cvt(in_proj_w,  w_in,   Ex * Cx);
    cvt(x_proj_w,   w_xp,   KXP * DIx);
    cvt(x_proj_w_b, w_xpb,  KXP * DIx);
    cvt(out_proj_w, w_out,  Cx * DIx);
    cvt(proj_w,     w_proj, Cx * Cx);

    const int nEl = Bx * DIx * Lx;           // 16,777,216
    const int gEl = nEl / 256;               // 65,536 blocks

    // 2) LN1 -> xn f16
    ln1_kernel<<<Bx * Lx, 256, 0, stream>>>(x, norm_g, norm_b, xn);

    // 3) in_proj GEMM -> xz (B, 1024, L)
    gemm_inproj_kernel<<<dim3(Lx / 16 / 8, Ex / 64, Bx), 256, 0, stream>>>(xn, w_in, xz);

    // 4) conv + SiLU (both directions)
    conv_silu_kernel<<<gEl, 256, 0, stream>>>(xz, conv_w,   conv_b,   xc_f, xcT_f, 0);
    conv_silu_kernel<<<gEl, 256, 0, stream>>>(xz, conv_w_b, conv_b_b, xc_r, xcT_r, 1);

    // 5) x_proj GEMMs -> x_dbl (B, 48, L)
    gemm_xproj_kernel<<<dim3(Lx / 16 / 8, 1, Bx), 256, 0, stream>>>(xcT_f, w_xp,  xdbl_f);
    gemm_xproj_kernel<<<dim3(Lx / 16 / 8, 1, Bx), 256, 0, stream>>>(xcT_r, w_xpb, xdbl_r);

    // 6) dt = softplus(x_dbl[:, :16] @ dt_w + dt_b)
    dt_kernel<<<gEl, 256, 0, stream>>>(xdbl_f, dt_proj_w,   dt_proj_b,   dt_f);
    dt_kernel<<<gEl, 256, 0, stream>>>(xdbl_r, dt_proj_w_b, dt_proj_b_b, dt_r);

    // 7) selective scans
    scan_kernel<<<dim3(DIx / 16, Bx), 256, 0, stream>>>(dt_f, xc_f, xdbl_f, A_log,   ys_f);
    scan_kernel<<<dim3(DIx / 16, Bx), 256, 0, stream>>>(dt_r, xc_r, xdbl_r, A_log_b, ys_r);

    // 8) combine directions + gate -> ycT f16 (B, L, 512)
    combine_kernel<<<gEl, 256, 0, stream>>>(ys_f, ys_r, xc_f, xc_r, xz, Dp, Dp_b, ycT);

    // 9) out_proj GEMM -> mamba_out (B, L, 256)
    gemm_outproj_kernel<<<dim3(Lx / 16 / 8, Cx / 64, Bx), 256, 0, stream>>>(ycT, w_out, mo);

    // 10) skip + LN2 -> xm f16
    ln2_kernel<<<Bx * Lx, 256, 0, stream>>>(mo, x, skip_scale, norm_g, norm_b, xm);

    // 11) final proj GEMM + bias -> out (B, 256, L)
    gemm_proj_kernel<<<dim3(Lx / 16 / 8, Cx / 64, Bx), 256, 0, stream>>>(xm, w_proj, proj_b, out);
}
